// Flow_Attention_Mask_68298569941074
// MI455X (gfx1250) — matrix operand order repaired and checked by reading, not compiled, not run on hardware
//
#include <hip/hip_runtime.h>
#include <hip/hip_bf16.h>
#include <stdint.h>

// ---------------------------------------------------------------------------
// Flow-Attention for MI455X (gfx1250, wave32, WMMA bf16 16x16x32, f32 accum)
// Big GEMMs: async global->LDS staged B (ASYNCcnt, double buffered), fully
// unrolled k-loop, 32x64 output tile per wave (8 in-place WMMA accumulators).
// ---------------------------------------------------------------------------

typedef __bf16 bf16_t;
typedef __attribute__((ext_vector_type(16))) __bf16 v16bf;
typedef __attribute__((ext_vector_type(8)))  __bf16 v8bf;
typedef __attribute__((ext_vector_type(8)))  float  v8f;

#define B_   8
#define N_   4096
#define C_   768
#define H_   12
#define D_   64
#define BH_  96
#define BN_  32768
#define EPS_ 1e-6f

__device__ __forceinline__ float bf2f(bf16_t b) {
  unsigned short s = __builtin_bit_cast(unsigned short, b);
  unsigned int u = ((unsigned int)s) << 16;
  return __builtin_bit_cast(float, u);
}
__device__ __forceinline__ bf16_t f2bf(float f) {
  unsigned int u = __builtin_bit_cast(unsigned int, f);
  unsigned int r = u + 0x7FFFu + ((u >> 16) & 1u);   // round-to-nearest-even
  unsigned short s = (unsigned short)(r >> 16);
  return __builtin_bit_cast(bf16_t, s);
}
__device__ __forceinline__ float sigmoidf_(float x) { return 1.0f / (1.0f + __expf(-x)); }

// A-operand (16x32 bf16): lane L<16 holds row M=L, K = {k0..k0+15};
// lane L>=16 holds row M=L-16, K = {k0+16..k0+31}.
__device__ __forceinline__ v16bf load_a16x32(const bf16_t* p, int sel) {
  union { v16bf v; v8bf h[2]; } u;
  u.h[0] = *reinterpret_cast<const v8bf*>(p + sel * 16);
  u.h[1] = *reinterpret_cast<const v8bf*>(p + sel * 16 + 8);
  return u.v;
}
// B-operand from pre-packed layout: 32 contiguous bytes per lane (global or LDS).
__device__ __forceinline__ v16bf load_b32(const bf16_t* p) {
  union { v16bf v; v8bf h[2]; } u;
  u.h[0] = *reinterpret_cast<const v8bf*>(p);
  u.h[1] = *reinterpret_cast<const v8bf*>(p + 8);
  return u.v;
}
__device__ __forceinline__ v8f wmma_bf16(v16bf a, v16bf b, v8f c) {
  return __builtin_amdgcn_wmma_f32_16x16x32_bf16(false, a, false, b, (short)0, c, false, false);
}

// Async global->LDS 16-byte copy (per lane), tracked with ASYNCcnt.
// Generic LDS pointer low 32 bits == LDS byte offset (ISA flat->LDS mapping).
__device__ __forceinline__ void async_ld_b128(const bf16_t* g, bf16_t* l) {
  unsigned long long ga = (unsigned long long)(uintptr_t)g;
  unsigned int la = (unsigned int)(uintptr_t)l;
  asm volatile("global_load_async_to_lds_b128 %0, %1, off" :: "v"(la), "v"(ga) : "memory");
}
__device__ __forceinline__ void wait_async_le2() {
  asm volatile("s_wait_asynccnt 0x2" ::: "memory");
}
__device__ __forceinline__ void wait_async_0() {
  asm volatile("s_wait_asynccnt 0x0" ::: "memory");
}

// --------------------------- setup / packing -------------------------------

__global__ void cvt_bf16_kernel(const float* __restrict__ src, bf16_t* __restrict__ dst, int n) {
  int i = blockIdx.x * blockDim.x + threadIdx.x;
  if (i < n) dst[i] = f2bf(src[i]);
}

// Pack a row-major [K x ncols] f32 matrix into WMMA-B bf16 tiles.
__global__ void pack_w_kernel(const float* __restrict__ src, bf16_t* __restrict__ dst,
                              int K, int ncols) {
  int idx = blockIdx.x * blockDim.x + threadIdx.x;
  int total = K * ncols;
  if (idx >= total) return;
  int nK = K >> 5;
  int t = idx & 15;
  int lane = (idx >> 4) & 31;
  int rest = idx >> 9;
  int kstep = rest % nK;
  int ct = rest / nK;
  int kk = kstep * 32 + (lane & 16) + t;
  int col = ct * 16 + (lane & 15);
  dst[idx] = f2bf(src[(size_t)kk * ncols + col]);
}

// --------------------------- QKV projection --------------------------------
// grid (256, 36), block 128. Each wave: 32 rows x 64 cols (2 row tiles,
// 8 accumulators). B tiles staged once per block into LDS (async, dbl-buffer).
__global__ void gemm_qkv_kernel(const bf16_t* __restrict__ xb, const bf16_t* __restrict__ wp,
                                const float* __restrict__ bias,
                                bf16_t* __restrict__ q, bf16_t* __restrict__ kT,
                                bf16_t* __restrict__ vp) {
  __shared__ bf16_t sb[2][4][512];           // 2 buffers x 4 tiles x 1KB
  const int lane = threadIdx.x & 31;
  const int wave = threadIdx.x >> 5;
  const int rowBase = (blockIdx.x * 4 + wave) * 32;
  const int cg = blockIdx.y;          // 0..35
  const int which = cg / H_;          // 0=q 1=k 2=v
  const int h = cg % H_;
  const int sel = lane >> 4;
  const int llo = lane & 15;

  // thread stages 32B of tile j==wave per k-step
  const bf16_t* wsrc = wp + ((size_t)((cg * 4 + wave) * 24) * 32 + lane) * 16;
  const bf16_t* arow0 = xb + (size_t)(rowBase + llo) * C_;
  const bf16_t* arow1 = xb + (size_t)(rowBase + 16 + llo) * C_;

  // prologue: stage ks=0 into buffer 0
  async_ld_b128(wsrc, &sb[0][wave][lane * 16]);
  async_ld_b128(wsrc + 8, &sb[0][wave][lane * 16 + 8]);

  v8f acc[2][4] = {};
#pragma unroll
  for (int ks = 0; ks < 24; ++ks) {
    const int cur = ks & 1;
    if (ks + 1 < 24) {
      const bf16_t* s = wsrc + (size_t)(ks + 1) * 512;
      async_ld_b128(s, &sb[cur ^ 1][wave][lane * 16]);
      async_ld_b128(s + 8, &sb[cur ^ 1][wave][lane * 16 + 8]);
      wait_async_le2();                 // my ks-loads done; ks+1 stays in flight
    } else {
      wait_async_0();
    }
    __syncthreads();                    // all waves' ks-loads visible in LDS
    v16bf a0 = load_a16x32(arow0 + ks * 32, sel);
    v16bf a1 = load_a16x32(arow1 + ks * 32, sel);
#pragma unroll
    for (int j = 0; j < 4; ++j) {
      v16bf b = load_b32(&sb[cur][j][lane * 16]);
      acc[0][j] = wmma_bf16(a0, b, acc[0][j]);
      acc[1][j] = wmma_bf16(a1, b, acc[1][j]);
    }
    __syncthreads();                    // buffer safe to overwrite next step
  }
#pragma unroll
  for (int rt = 0; rt < 2; ++rt) {
#pragma unroll
    for (int j = 0; j < 4; ++j) {
#pragma unroll
      for (int r = 0; r < 8; ++r) {
        int m = rowBase + rt * 16 + r + sel * 8;
        int bb = m >> 12;
        int n = m & (N_ - 1);
        int d = j * 16 + llo;
        int col = cg * 64 + d;
        float val = acc[rt][j][r] + bias[col];
        int bh = bb * H_ + h;
        if (which == 0) {
          val = sigmoidf_(val);
          q[((size_t)bh * N_ + n) * D_ + d] = f2bf(val);
        } else if (which == 1) {
          val = sigmoidf_(val);
          kT[((size_t)bh * D_ + d) * N_ + n] = f2bf(val);
        } else {
          // v -> packed WMMA-B layout [bh][ct2][kstep][lane2][t]
          int ct2 = d >> 4;
          int kstep = n >> 5;
          int lane2 = (d & 15) + (n & 16);
          int t = n & 15;
          vp[((((size_t)bh * 4 + ct2) * 128 + kstep) * 32 + lane2) * 16 + t] = f2bf(val);
        }
      }
    }
  }
}

// --------------------------- flow statistics -------------------------------

// grid 96, block 64 (d = tid): qsum/ksum over N
__global__ void colsums_kernel(const bf16_t* __restrict__ q, const bf16_t* __restrict__ kT,
                               float* __restrict__ qsum, float* __restrict__ ksum) {
  int bh = blockIdx.x;
  int d = threadIdx.x;
  const bf16_t* kr = kT + ((size_t)bh * D_ + d) * N_;
  float sq = 0.f, sk = 0.f;
  for (int n = 0; n < N_; ++n) {
    sq += bf2f(q[((size_t)bh * N_ + n) * D_ + d]);
    sk += bf2f(kr[n]);
  }
  qsum[bh * D_ + d] = sq;
  ksum[bh * D_ + d] = sk;
}

// grid 1536, block 256: sink_incoming / source_outgoing per (bh,n)
__global__ void flow_norm_kernel(const bf16_t* __restrict__ q, const bf16_t* __restrict__ kT,
                                 const float* __restrict__ qsum, const float* __restrict__ ksum,
                                 float* __restrict__ si, float* __restrict__ so) {
  int bh = blockIdx.x >> 4;
  int n = ((blockIdx.x & 15) << 8) | threadIdx.x;
  const bf16_t* qrow = q + ((size_t)bh * N_ + n) * D_;
  float dsink = 0.f, dsrc = 0.f;
  for (int d = 0; d < D_; ++d) {
    float qd = bf2f(qrow[d]) + EPS_;
    float kd = bf2f(kT[((size_t)bh * D_ + d) * N_ + n]) + EPS_;
    dsink += qd * (ksum[bh * D_ + d] + EPS_);
    dsrc  += kd * (qsum[bh * D_ + d] + EPS_);
  }
  si[(size_t)bh * N_ + n] = 1.0f / (dsink + EPS_);
  so[(size_t)bh * N_ + n] = 1.0f / (dsrc + EPS_);
}

// grid 96, block 64: weighted column sums (+EPS)
__global__ void wcolsum_kernel(const bf16_t* __restrict__ q, const bf16_t* __restrict__ kT,
                               const float* __restrict__ si, const float* __restrict__ so,
                               float* __restrict__ qwsum, float* __restrict__ kwsum) {
  int bh = blockIdx.x;
  int d = threadIdx.x;
  const bf16_t* kr = kT + ((size_t)bh * D_ + d) * N_;
  float qw = 0.f, kw = 0.f;
  for (int n = 0; n < N_; ++n) {
    qw += bf2f(q[((size_t)bh * N_ + n) * D_ + d]) * si[(size_t)bh * N_ + n];
    kw += bf2f(kr[n]) * so[(size_t)bh * N_ + n];
  }
  qwsum[bh * D_ + d] = qw + EPS_;
  kwsum[bh * D_ + d] = kw + EPS_;
}

// grid 1536, block 256: conserved sink/source, sigmoid allocation, exp for softmax
__global__ void conserve_kernel(const bf16_t* __restrict__ q, const bf16_t* __restrict__ kT,
                                const float* __restrict__ qwsum, const float* __restrict__ kwsum,
                                const float* __restrict__ si,
                                float* __restrict__ scaleArr, float* __restrict__ eArr,
                                float* __restrict__ epart) {
  __shared__ float red[256];
  int bh = blockIdx.x >> 4;
  int n = ((blockIdx.x & 15) << 8) | threadIdx.x;
  const bf16_t* qrow = q + ((size_t)bh * N_ + n) * D_;
  float csink = EPS_, csrc = EPS_;
  for (int d = 0; d < D_; ++d) {
    csink += (bf2f(qrow[d]) + EPS_) * kwsum[bh * D_ + d];
    csrc  += (bf2f(kT[((size_t)bh * D_ + d) * N_ + n]) + EPS_) * qwsum[bh * D_ + d];
  }
  csrc = fminf(1.0f, fmaxf(-1.0f, csrc));
  float alloc = sigmoidf_(csink);
  scaleArr[(size_t)bh * N_ + n] = si[(size_t)bh * N_ + n] * alloc;
  float e = __expf(csrc);
  eArr[(size_t)bh * N_ + n] = e;
  red[threadIdx.x] = e;
  __syncthreads();
  for (int s = 128; s > 0; s >>= 1) {
    if (threadIdx.x < s) red[threadIdx.x] += red[threadIdx.x + s];
    __syncthreads();
  }
  if (threadIdx.x == 0) epart[bh * 16 + (blockIdx.x & 15)] = red[0];
}

// 1 block: esum[bh] = sum of 16 partials (deterministic)
__global__ void reduce_esum_kernel(const float* __restrict__ epart, float* __restrict__ esum) {
  int bh = threadIdx.x;
  if (bh < BH_) {
    float s = 0.f;
    for (int i = 0; i < 16; ++i) s += epart[bh * 16 + i];
    esum[bh] = s;
  }
}

// scale packed v in place by source_competition = e / esum * N
__global__ void vc_scale_kernel(bf16_t* __restrict__ vp, const float* __restrict__ eArr,
                                const float* __restrict__ esum, int total) {
  int idx = blockIdx.x * blockDim.x + threadIdx.x;
  if (idx >= total) return;
  int t = idx & 15;
  int lane = (idx >> 4) & 31;
  int rest = idx >> 9;
  int kstep = rest & 127;
  rest >>= 7;
  int bh = rest >> 2;
  int n = kstep * 32 + (lane & 16) + t;
  float comp = eArr[(size_t)bh * N_ + n] * ((float)N_ / esum[bh]);
  vp[idx] = f2bf(bf2f(vp[idx]) * comp);
}

// --------------------------- kv = k^T (v*comp) -----------------------------
// grid (96, 8 chunks of N), block 128 (wave = d-tile). Partial kv per chunk.
__global__ void gemm_kv_kernel(const bf16_t* __restrict__ kT, const bf16_t* __restrict__ vp,
                               float* __restrict__ kvPart) {
  const int lane = threadIdx.x & 31;
  const int d0 = threadIdx.x >> 5;
  const int bh = blockIdx.x;
  const int chunk = blockIdx.y;
  const int sel = lane >> 4;
  const int llo = lane & 15;

  const bf16_t* arow = kT + ((size_t)bh * D_ + d0 * 16 + llo) * N_ + chunk * 512;
  v8f acc[4] = {};
  for (int ks = 0; ks < 16; ++ks) {
    v16bf a = load_a16x32(arow + ks * 32, sel);
#pragma unroll
    for (int ct = 0; ct < 4; ++ct) {
      v16bf b = load_b32(vp + ((((size_t)bh * 4 + ct) * 128 + chunk * 16 + ks) * 32 + lane) * 16);
      acc[ct] = wmma_bf16(a, b, acc[ct]);
    }
  }
#pragma unroll
  for (int ct = 0; ct < 4; ++ct) {
#pragma unroll
    for (int r = 0; r < 8; ++r) {
      int dI = d0 * 16 + r + sel * 8;
      int e = ct * 16 + llo;
      kvPart[((size_t)(chunk * BH_ + bh) * D_ + dI) * D_ + e] = acc[ct][r];
    }
  }
}

// Deterministically reduce 8 partials and pack kv into WMMA-B layout (bf16).
__global__ void kv_pack_kernel(const float* __restrict__ kvPart, bf16_t* __restrict__ kvp,
                               int total) {
  int idx = blockIdx.x * blockDim.x + threadIdx.x;
  if (idx >= total) return;
  int t = idx & 15;
  int lane = (idx >> 4) & 31;
  int rest = idx >> 9;
  int ks = rest & 1;
  rest >>= 1;
  int ct = rest & 3;
  int bh = rest >> 2;
  int kk = ks * 32 + (lane & 16) + t;
  int e = ct * 16 + (lane & 15);
  float s = 0.f;
  for (int c = 0; c < 8; ++c) s += kvPart[((size_t)(c * BH_ + bh) * D_ + kk) * D_ + e];
  kvp[idx] = f2bf(s);
}

// --------------------------- x_update = q @ kv -----------------------------
__global__ void gemm_xupd_kernel(const bf16_t* __restrict__ q, const bf16_t* __restrict__ kvp,
                                 const float* __restrict__ scaleArr, bf16_t* __restrict__ attnb) {
  const int lane = threadIdx.x & 31;
  const int wave = threadIdx.x >> 5;
  const int rt = blockIdx.x * 4 + wave;
  const int bh = rt >> 8;
  const int n0 = (rt & 255) * 16;
  const int sel = lane >> 4;
  const int llo = lane & 15;

  const bf16_t* arow = q + ((size_t)bh * N_ + n0 + llo) * D_;
  v8f acc[4] = {};
#pragma unroll
  for (int ks = 0; ks < 2; ++ks) {
    v16bf a = load_a16x32(arow + ks * 32, sel);
#pragma unroll
    for (int ct = 0; ct < 4; ++ct) {
      v16bf b = load_b32(kvp + ((((size_t)bh * 4 + ct) * 2 + ks) * 32 + lane) * 16);
      acc[ct] = wmma_bf16(a, b, acc[ct]);
    }
  }
  int bb = bh / H_, h = bh % H_;
#pragma unroll
  for (int ct = 0; ct < 4; ++ct) {
#pragma unroll
    for (int r = 0; r < 8; ++r) {
      int n = n0 + r + sel * 8;
      float s = scaleArr[(size_t)bh * N_ + n];
      int col = ct * 16 + llo;
      attnb[((size_t)(bb * N_ + n)) * C_ + h * D_ + col] = f2bf(acc[ct][r] * s);
    }
  }
}

// --------------------------- output projection -----------------------------
// grid (256, 12), block 128; 32x64 tile per wave; async-LDS staged B.
__global__ void gemm_proj_kernel(const bf16_t* __restrict__ attnb, const bf16_t* __restrict__ wp,
                                 const float* __restrict__ bias, float* __restrict__ out) {
  __shared__ bf16_t sb[2][4][512];
  const int lane = threadIdx.x & 31;
  const int wave = threadIdx.x >> 5;
  const int rowBase = (blockIdx.x * 4 + wave) * 32;
  const int cg = blockIdx.y;
  const int sel = lane >> 4;
  const int llo = lane & 15;

  const bf16_t* wsrc = wp + ((size_t)((cg * 4 + wave) * 24) * 32 + lane) * 16;
  const bf16_t* arow0 = attnb + (size_t)(rowBase + llo) * C_;
  const bf16_t* arow1 = attnb + (size_t)(rowBase + 16 + llo) * C_;

  async_ld_b128(wsrc, &sb[0][wave][lane * 16]);
  async_ld_b128(wsrc + 8, &sb[0][wave][lane * 16 + 8]);

  v8f acc[2][4] = {};
#pragma unroll
  for (int ks = 0; ks < 24; ++ks) {
    const int cur = ks & 1;
    if (ks + 1 < 24) {
      const bf16_t* s = wsrc + (size_t)(ks + 1) * 512;
      async_ld_b128(s, &sb[cur ^ 1][wave][lane * 16]);
      async_ld_b128(s + 8, &sb[cur ^ 1][wave][lane * 16 + 8]);
      wait_async_le2();
    } else {
      wait_async_0();
    }
    __syncthreads();
    v16bf a0 = load_a16x32(arow0 + ks * 32, sel);
    v16bf a1 = load_a16x32(arow1 + ks * 32, sel);
#pragma unroll
    for (int j = 0; j < 4; ++j) {
      v16bf b = load_b32(&sb[cur][j][lane * 16]);
      acc[0][j] = wmma_bf16(a0, b, acc[0][j]);
      acc[1][j] = wmma_bf16(a1, b, acc[1][j]);
    }
    __syncthreads();
  }
#pragma unroll
  for (int rt = 0; rt < 2; ++rt) {
#pragma unroll
    for (int j = 0; j < 4; ++j) {
#pragma unroll
      for (int r = 0; r < 8; ++r) {
        int m = rowBase + rt * 16 + r + sel * 8;
        int col = cg * 64 + j * 16 + llo;
        float val = acc[rt][j][r] + bias[col];
        out[(size_t)m * C_ + col] = sigmoidf_(2.0f * val);
      }
    }
  }
}

// ---------------------------------------------------------------------------

extern "C" void kernel_launch(void* const* d_in, const int* in_sizes, int n_in,
                              void* d_out, int out_size, void* d_ws, size_t ws_size,
                              hipStream_t stream) {
  (void)in_sizes; (void)n_in; (void)out_size; (void)ws_size;
  const float* x     = (const float*)d_in[0];
  const float* Wqkv  = (const float*)d_in[1];
  const float* bqkv  = (const float*)d_in[2];
  const float* Wproj = (const float*)d_in[3];
  const float* bproj = (const float*)d_in[4];
  float* out = (float*)d_out;

  char* w = (char*)d_ws;
  auto take = [&](size_t bytes) -> char* {
    char* p = w;
    w += (bytes + 255) & ~(size_t)255;
    return p;
  };

  bf16_t* xb     = (bf16_t*)take((size_t)BN_ * C_ * 2);
  bf16_t* wqkvP  = (bf16_t*)take((size_t)C_ * 3 * C_ * 2);
  bf16_t* wprojP = (bf16_t*)take((size_t)C_ * C_ * 2);
  bf16_t* qArr   = (bf16_t*)take((size_t)BH_ * N_ * D_ * 2);
  bf16_t* kT     = (bf16_t*)take((size_t)BH_ * D_ * N_ * 2);
  bf16_t* vP     = (bf16_t*)take((size_t)BH_ * N_ * D_ * 2);
  bf16_t* attnb  = (bf16_t*)take((size_t)BN_ * C_ * 2);
  float*  qsum   = (float*)take((size_t)BH_ * D_ * 4);
  float*  ksum   = (float*)take((size_t)BH_ * D_ * 4);
  float*  qwsum  = (float*)take((size_t)BH_ * D_ * 4);
  float*  kwsum  = (float*)take((size_t)BH_ * D_ * 4);
  float*  si     = (float*)take((size_t)BH_ * N_ * 4);
  float*  so     = (float*)take((size_t)BH_ * N_ * 4);
  float*  scaleA = (float*)take((size_t)BH_ * N_ * 4);
  float*  eArr   = (float*)take((size_t)BH_ * N_ * 4);
  float*  epart  = (float*)take((size_t)BH_ * 16 * 4);
  float*  esum   = (float*)take((size_t)BH_ * 4);
  float*  kvPart = (float*)take((size_t)8 * BH_ * D_ * D_ * 4);
  bf16_t* kvp    = (bf16_t*)take((size_t)BH_ * 4 * 2 * 32 * 16 * 2);

  cvt_bf16_kernel<<<(BN_ * C_) / 256, 256, 0, stream>>>(x, xb, BN_ * C_);
  pack_w_kernel<<<(C_ * 3 * C_) / 256, 256, 0, stream>>>(Wqkv, wqkvP, C_, 3 * C_);
  pack_w_kernel<<<(C_ * C_) / 256, 256, 0, stream>>>(Wproj, wprojP, C_, C_);

  gemm_qkv_kernel<<<dim3(BN_ / 128, 36), 128, 0, stream>>>(xb, wqkvP, bqkv, qArr, kT, vP);

  colsums_kernel<<<BH_, D_, 0, stream>>>(qArr, kT, qsum, ksum);
  flow_norm_kernel<<<BH_ * 16, 256, 0, stream>>>(qArr, kT, qsum, ksum, si, so);
  wcolsum_kernel<<<BH_, D_, 0, stream>>>(qArr, kT, si, so, qwsum, kwsum);
  conserve_kernel<<<BH_ * 16, 256, 0, stream>>>(qArr, kT, qwsum, kwsum, si, scaleA, eArr, epart);
  reduce_esum_kernel<<<1, 128, 0, stream>>>(epart, esum);
  vc_scale_kernel<<<(BH_ * N_ * D_) / 256, 256, 0, stream>>>(vP, eArr, esum, BH_ * N_ * D_);

  gemm_kv_kernel<<<dim3(BH_, 8), 128, 0, stream>>>(kT, vP, kvPart);
  kv_pack_kernel<<<(BH_ * 4096) / 256, 256, 0, stream>>>(kvPart, kvp, BH_ * 4096);

  gemm_xupd_kernel<<<(BH_ * 256) / 4, 128, 0, stream>>>(qArr, kvp, scaleA, attnb);

  gemm_proj_kernel<<<dim3(BN_ / 128, C_ / 64), 128, 0, stream>>>(attnb, wprojP, bproj, out);
}
